// LSTM_62809601737255
// MI455X (gfx1250) — matrix operand order repaired and checked by reading, not compiled, not run on hardware
//
#include <hip/hip_runtime.h>
#include <hip/hip_bf16.h>

typedef __attribute__((ext_vector_type(16))) _Float16 v16h;
typedef __attribute__((ext_vector_type(8)))  float    v8f;

#define C4   4
#define HID  256
#define INP  256
#define G4H  1024          // 4*HID
#define KTOT 512           // INP + HID
#define BMAX 64            // padded batch rows (actual B=48)
#define AROW 520           // LDS A-panel row stride in halfs (512 + 8 pad -> 1040 B)

// ---------------------------------------------------------------- metadata
__global__ void meta_kernel(const int* __restrict__ bs, int T,
                            int* __restrict__ off, int* __restrict__ len,
                            int* __restrict__ rev) {
    const int tid = threadIdx.x;
    if (tid == 0) {
        int a = 0;
        for (int t = 0; t < T; ++t) { off[t] = a; a += bs[t]; }
        off[T] = a;
    }
    __syncthreads();
    const int B = bs[0];
    if (tid < B) {
        int cnt = 0;
        for (int t = 0; t < T; ++t) cnt += (bs[t] > tid) ? 1 : 0;
        len[tid] = cnt;
    }
    __syncthreads();
    for (int idx = tid; idx < T * B; idx += blockDim.x) {
        const int t = idx / B, b = idx % B;
        if (b < bs[t]) rev[off[t] + b] = off[len[b] - 1 - t] + b;
    }
}

// ---------------------------------------------------------------- bias fuse
__global__ void bias_kernel(const float* __restrict__ bih, const float* __restrict__ bhh,
                            const float* __restrict__ bihR, const float* __restrict__ bhhR,
                            float* __restrict__ bias) {
    const int id = blockIdx.x * blockDim.x + threadIdx.x;          // 8*1024
    if (id >= 8 * G4H) return;
    const int c2 = id >> 10, g = id & (G4H - 1);
    bias[id] = (c2 < C4) ? (bih[c2 * G4H + g] + bhh[c2 * G4H + g])
                         : (bihR[(c2 - C4) * G4H + g] + bhhR[(c2 - C4) * G4H + g]);
}

// ------------------------------------------------- pack weights -> f16 B-fragments
// Layout: [c2(8)][gtile(64)][kstep(16)][lane(32)][16 halfs]; combined K = [I | H].
// Per (gtile,kstep): 1 KB block; per gtile: 16 KB; per c2: 1 MB.
__global__ void pack_w_kernel(const float* __restrict__ Wih, const float* __restrict__ Whh,
                              const float* __restrict__ WihR, const float* __restrict__ WhhR,
                              unsigned short* __restrict__ Wp_) {
    const int id = blockIdx.x * blockDim.x + threadIdx.x;          // 8*64*16*32
    if (id >= 8 * 64 * 16 * 32) return;
    const int lane  = id & 31;
    const int s     = (id >> 5) & 15;
    const int gtile = (id >> 9) & 63;
    const int c2    = id >> 15;
    const int g  = gtile * 16 + (lane & 15);
    const int k0 = s * 32 + (lane >> 4) * 16;                       // 16 contiguous K
    const float* src;
    if (k0 < INP) {
        const float* W = (c2 < C4) ? Wih  + ((size_t)c2 * G4H + g) * INP
                                   : WihR + ((size_t)(c2 - C4) * G4H + g) * INP;
        src = W + k0;
    } else {
        const float* W = (c2 < C4) ? Whh  + ((size_t)c2 * G4H + g) * HID
                                   : WhhR + ((size_t)(c2 - C4) * G4H + g) * HID;
        src = W + (k0 - INP);
    }
    v16h v;
    #pragma unroll
    for (int i = 0; i < 16; ++i) v[i] = (_Float16)src[i];
    *(v16h*)((_Float16*)Wp_ + (size_t)id * 16) = v;
}

// ---------------------------------------------------------------- zero state
__global__ void init_state(unsigned short* __restrict__ h, float* __restrict__ c, int n) {
    const int i = blockIdx.x * blockDim.x + threadIdx.x;
    if (i < n) { h[i] = 0; c[i] = 0.f; }
}

// ---------------------------------------------------------------- LSTM step
// grid: (8 dir-channels, 4 row-tiles) x 256 threads (8 waves).
// Wave w owns gate columns [32w, 32w+32) in each of the 4 gate quarters.
__global__ __launch_bounds__(256) void step_kernel(
        int t, const float* __restrict__ data, const int* __restrict__ bs,
        const int* __restrict__ off, const int* __restrict__ rev,
        const float* __restrict__ bias, const unsigned short* __restrict__ Wp_,
        unsigned short* __restrict__ hst_, float* __restrict__ cst,
        float* __restrict__ out) {
    const int c2    = blockIdx.x;          // 0..7 (fwd c / bwd c)
    const int mtile = blockIdx.y;          // 0..3 -> rows [16*mtile, 16*mtile+16)
    const int bs_t  = bs[t];
    if (mtile * 16 >= bs_t) return;
    const int off_t = off[t];
    const int w    = threadIdx.x >> 5;
    const int lane = threadIdx.x & 31;
    const int hi   = lane >> 4;
    const int mrow = lane & 15;
    _Float16* hst = (_Float16*)hst_;

    // B fragments: single per-wave byte base, all tile/kstep offsets are immediates
    const char* wb = (const char*)Wp_
                   + (size_t)c2 * (1u << 20)      // 1 MB per channel
                   + (size_t)(2 * w) * 16384      // wave's first gtile
                   + (size_t)lane * 32;           // lane stride within 1 KB block

    // load the 8 B fragments of kstep s into dst (16 x b128, immediate offsets)
    auto loadB = [&](v16h* dst, int s) {
        const char* pb = wb + s * 1024;
        #pragma unroll
        for (int ti = 0; ti < 8; ++ti) {
            // quarter q = ti>>1 (256 KB apart), pair pp = ti&1 (16 KB apart)
            dst[ti] = *(const v16h*)(pb + (ti >> 1) * 262144 + (ti & 1) * 16384);
        }
    };

    v16h bufA[8], bufB[8];
    loadB(bufA, 0);                                // in flight across staging+barrier

    // ---- stage the 16x512 f16 A panel in LDS (x part converted, h part copied)
    __shared__ __align__(16) unsigned short aSh[16 * AROW];
    #pragma unroll
    for (int i = 0; i < 4; ++i) {                  // x part: 1024 float4 chunks
        const int q    = threadIdx.x + i * 256;
        const int row  = q >> 6;                   // 64 chunks of 4 floats per row
        const int col4 = q & 63;
        const int br = mtile * 16 + row;
        const int bc = br < bs_t ? br : (bs_t - 1);
        const int nt = off_t + bc;
        const int xt = (c2 < C4) ? nt : rev[nt];
        const float4 f = *(const float4*)(data + ((size_t)xt * C4 + (c2 & 3)) * INP + col4 * 4);
        union { _Float16 h[4]; uint2 u; } pk;
        pk.h[0] = (_Float16)f.x; pk.h[1] = (_Float16)f.y;
        pk.h[2] = (_Float16)f.z; pk.h[3] = (_Float16)f.w;
        *(uint2*)(aSh + row * AROW + col4 * 4) = pk.u;
    }
    #pragma unroll
    for (int i = 0; i < 2; ++i) {                  // h part: 512 chunks of 8 halfs
        const int cid = threadIdx.x + i * 256;
        const int row = cid >> 5;
        const int c8  = cid & 31;
        const int br  = mtile * 16 + row;
        const uint4 u = *(const uint4*)(hst + ((size_t)br * 8 + c2) * HID + c8 * 8);
        *(uint4*)(aSh + row * AROW + 256 + c8 * 8) = u;
    }
    __syncthreads();

    // A fragment for kstep s straight from LDS (contiguous K order k = 16*hi + e, matching packed B, DScnt only)
    const unsigned short* aBase = aSh + mrow * AROW + hi * 16;
    auto ldsA = [&](int s) -> v16h {
        v16h a;
        const unsigned short* ap = aBase + s * 32;
        ((uint4*)&a)[0] = *(const uint4*)(ap);
        ((uint4*)&a)[1] = *(const uint4*)(ap + 8);
        return a;
    };

    const v8f vz = {};
    v8f acc[8];
    #pragma unroll
    for (int i = 0; i < 8; ++i) acc[i] = vz;

    #pragma unroll 1
    for (int s = 0; s < 16; s += 2) {              // K = 512 in chunks of 32
        loadB(bufB, s + 1);                        // prefetch next kstep (stays in flight)
        {
            const v16h a = ldsA(s);
            #pragma unroll
            for (int ti = 0; ti < 8; ++ti)
                acc[ti] = __builtin_amdgcn_wmma_f32_16x16x32_f16(
                    false, a, false, bufA[ti], (short)0, acc[ti], false, false);
        }
        const int s2 = (s + 2 < 16) ? (s + 2) : 15;  // clamped redundant reload, no branch
        loadB(bufA, s2);
        {
            const v16h a = ldsA(s + 1);
            #pragma unroll
            for (int ti = 0; ti < 8; ++ti)
                acc[ti] = __builtin_amdgcn_wmma_f32_16x16x32_f16(
                    false, a, false, bufB[ti], (short)0, acc[ti], false, false);
        }
    }

    // ---- cell update: this wave holds i/f/g/o for columns j = 32w + pp*16 + mrow
    const int jb = w * 32 + mrow;
    #pragma unroll
    for (int pp = 0; pp < 2; ++pp) {
        const int j = jb + pp * 16;
        const float bi  = bias[c2 * G4H + j];
        const float bfv = bias[c2 * G4H + 256 + j];
        const float bg  = bias[c2 * G4H + 512 + j];
        const float bo  = bias[c2 * G4H + 768 + j];
        #pragma unroll
        for (int r = 0; r < 8; ++r) {
            const int b = mtile * 16 + hi * 8 + r;     // C/D layout: lane<16 -> M=r, else 8+r
            if (b < bs_t) {
                const size_t srow = ((size_t)b * 8 + c2) * HID + j;
                const float cold = cst[srow];
                const float gi = 1.f / (1.f + __expf(-(acc[0 + pp][r] + bi)));
                const float gf = 1.f / (1.f + __expf(-(acc[2 + pp][r] + bfv)));
                const float gg = tanhf(acc[4 + pp][r] + bg);
                const float go = 1.f / (1.f + __expf(-(acc[6 + pp][r] + bo)));
                const float cn = gi * gg + gf * cold;
                const float hn = go * tanhf(cn);
                cst[srow] = cn;
                hst[srow] = (_Float16)hn;
                const int n = off_t + b;
                if (c2 < C4) out[((size_t)n * C4 + c2) * (2 * HID) + j] = hn;
                else out[((size_t)rev[n] * C4 + (c2 - C4)) * (2 * HID) + HID + j] = hn;
            }
        }
    }
}

// ---------------------------------------------------------------- launcher
extern "C" void kernel_launch(void* const* d_in, const int* in_sizes, int n_in,
                              void* d_out, int out_size, void* d_ws, size_t ws_size,
                              hipStream_t stream) {
    const int T = in_sizes[1];
    const int N = in_sizes[0] / (C4 * INP);

    const float* data = (const float*)d_in[0];
    const int*   bs   = (const int*)d_in[1];
    const float* Wih  = (const float*)d_in[2];
    const float* Whh  = (const float*)d_in[3];
    const float* bih  = (const float*)d_in[4];
    const float* bhh  = (const float*)d_in[5];
    const float* WihR = (const float*)d_in[6];
    const float* WhhR = (const float*)d_in[7];
    const float* bihR = (const float*)d_in[8];
    const float* bhhR = (const float*)d_in[9];

    char* p = (char*)d_ws;
    auto carve = [&](size_t bytes) {
        char* r = p; p += (bytes + 255) & ~(size_t)255; return r;
    };
    int*            off   = (int*)carve(sizeof(int) * (size_t)(T + 1));
    int*            len   = (int*)carve(sizeof(int) * BMAX);
    int*            rev   = (int*)carve(sizeof(int) * (size_t)N);
    float*          biasc = (float*)carve(sizeof(float) * 8 * G4H);
    unsigned short* Wp    = (unsigned short*)carve(2ull * 8 * 64 * 16 * 32 * 16);
    unsigned short* hst   = (unsigned short*)carve(2ull * BMAX * 8 * HID);
    float*          cst   = (float*)carve(4ull * BMAX * 8 * HID);
    (void)ws_size; (void)n_in; (void)out_size;

    meta_kernel<<<1, 256, 0, stream>>>(bs, T, off, len, rev);
    bias_kernel<<<(8 * G4H + 255) / 256, 256, 0, stream>>>(bih, bhh, bihR, bhhR, biasc);
    pack_w_kernel<<<(8 * 64 * 16 * 32) / 256, 256, 0, stream>>>(Wih, Whh, WihR, WhhR, Wp);
    init_state<<<(BMAX * 8 * HID + 255) / 256, 256, 0, stream>>>(hst, cst, BMAX * 8 * HID);

    for (int t = 0; t < T; ++t) {
        step_kernel<<<dim3(8, 4), 256, 0, stream>>>(
            t, data, bs, off, rev, biasc, Wp, hst, cst, (float*)d_out);
    }
}
